// FeedForwardQuantum_65481071396684
// MI455X (gfx1250) — matrix operand order repaired and checked by reading, not compiled, not run on hardware
//
#include <hip/hip_runtime.h>

// Problem constants (from reference): B=8, S=4096, E=256, NQ=8, F=1024
#define M_TOTAL 32768
#define E_DIM   256
#define NQ_DIM  8
#define F_DIM   1024

typedef __attribute__((ext_vector_type(16))) _Float16 v16h;
typedef __attribute__((ext_vector_type(8)))  _Float16 v8h;
typedef __attribute__((ext_vector_type(8)))  float    v8f;

// -------------------------------------------------------------------------
// Pack W1 (with cos(theta) folded in) into WMMA B-fragment layout, f16.
// B-matrix 32x16 f16 fragment: lanes 0-15 elems 0..7 hold K=0..7 (N=lane),
// every other element (lanes 16-31 too) is 0. K >= 8 is zero padding (NQ=8).
// w1p index: (f_tile*32 + lane)*16 halves, f_tile in [0,64).
// -------------------------------------------------------------------------
__global__ void pack_w1_kernel(const float* __restrict__ W1,
                               const float* __restrict__ theta,
                               _Float16* __restrict__ w1p) {
  int t = blockIdx.x * blockDim.x + threadIdx.x;   // 0..2047
  if (t >= 64 * 32) return;
  int lane   = t & 31;
  int f_tile = t >> 5;
  _Float16 h[16];
#pragma unroll
  for (int j = 0; j < 16; ++j) h[j] = (_Float16)0.0f;
  if (lane < 16) {
    int n = f_tile * 16 + lane;           // row of W1 [F, NQ]
#pragma unroll
    for (int j = 0; j < NQ_DIM; ++j)
      h[j] = (_Float16)(W1[n * NQ_DIM + j] * cosf(theta[j]));
  }
  _Float16* dst = w1p + t * 16;
#pragma unroll
  for (int j = 0; j < 16; ++j) dst[j] = h[j];
}

// -------------------------------------------------------------------------
// Pack W2 [E, F] into WMMA B-fragment layout, f16.
// Fragment (n_tile, k_step): lane<16 -> N=lane, K=k_step*32+{0..7,16..23}
//                            lane>=16 -> N=lane-16, K=k_step*32+{8..15,24..31}
// w2p index: ((n_tile*32 + k_step)*32 + lane)*16 halves.
// -------------------------------------------------------------------------
__global__ void pack_w2_kernel(const float* __restrict__ W2,
                               _Float16* __restrict__ w2p) {
  int t = blockIdx.x * blockDim.x + threadIdx.x;   // 0..16383
  if (t >= 16 * 32 * 32) return;
  int lane   = t & 31;
  int k_step = (t >> 5) & 31;
  int n_tile = t >> 10;
  int n  = n_tile * 16 + (lane & 15);               // row of W2 (E index)
  int kk = k_step * 32 + ((lane >> 4) << 3);        // F index base
  _Float16* dst = w2p + t * 16;
#pragma unroll
  for (int j = 0; j < 16; ++j)
    dst[j] = (_Float16)(W2[n * F_DIM + kk + j + ((j >> 3) << 3)]);
}

// -------------------------------------------------------------------------
// Fused main kernel.  Grid: 512 workgroups x 128 threads (4 waves, wave32).
// Each workgroup owns 64 tokens.  Stage A: per-wave WMMA q@W1p -> relu -> h
// (f16) staged in LDS.  Stage B: waves partition N (4 n-tiles each),
// accumulate 16 output tiles over K=1024 with v_wmma_f32_16x16x32_f16.
// LDS row stride 1032 halves (2064 B) => +4 banks/row, conflict-free b128.
// -------------------------------------------------------------------------
#define ROWSTRIDE_H 1032
#define LDS_BYTES  (64 * ROWSTRIDE_H * 2)   // 132096

__global__ __launch_bounds__(128, 1)
void ffq_main_kernel(const float* __restrict__ x,
                     const _Float16* __restrict__ w1p,
                     const _Float16* __restrict__ w2p,
                     float* __restrict__ out) {
  extern __shared__ _Float16 hsm[];   // 64 rows x ROWSTRIDE_H halves

  const int tid    = threadIdx.x;
  const int lane   = tid & 31;
  const int wave   = tid >> 5;
  const int lane15 = lane & 15;
  const int hiHalf = lane >> 4;       // 0: lanes 0-15, 1: lanes 16-31
  const int wgBase = blockIdx.x * 64; // first token row of this workgroup

  v8f zero8;
#pragma unroll
  for (int r = 0; r < 8; ++r) zero8[r] = 0.0f;

  // ---------------- Stage A: h = relu( cos(x)*cos(theta) @ W1^T ) ----------
  // A-fragment (16x32 f16): lane<16 -> M=lane, elems 0..7 = K0..7 (real q),
  // everything else zero (K>=8 padded; lanes>=16 cover K=8..15,24..31 -> 0).
  v16h aq;
#pragma unroll
  for (int j = 0; j < 16; ++j) aq[j] = (_Float16)0.0f;
  if (lane < 16) {
    int m = wgBase + wave * 16 + lane;
    const float* xr = x + (size_t)m * E_DIM;
#pragma unroll
    for (int j = 0; j < NQ_DIM; ++j)
      aq[j] = (_Float16)cosf(xr[j]);
  }

  const int myRow0 = wave * 16 + hiHalf * 8;   // D-tile: lane<16 M=r, else M=r+8
#pragma unroll 1
  for (int ft = 0; ft < 64; ++ft) {
    v16h b1 = *(const v16h*)(w1p + (ft * 32 + lane) * 16);
    v8f d = __builtin_amdgcn_wmma_f32_16x16x32_f16(
        false, aq, false, b1, (short)0, zero8, false, false);
    int col = ft * 16 + lane15;
#pragma unroll
    for (int r = 0; r < 8; ++r) {
      float v = d[r];
      v = v > 0.0f ? v : 0.0f;                 // ReLU
      hsm[(myRow0 + r) * ROWSTRIDE_H + col] = (_Float16)v;
    }
  }
  __syncthreads();

  // ---------------- Stage B: out = h @ W2^T --------------------------------
  v8f acc[4][4];
#pragma unroll
  for (int mt = 0; mt < 4; ++mt)
#pragma unroll
    for (int j = 0; j < 4; ++j) acc[mt][j] = zero8;

#pragma unroll 1
  for (int ks = 0; ks < 32; ++ks) {
    const int klo = ks * 32;                   // F-index base of this K slab
    // B fragments for this wave's 4 n-tiles (one contiguous 1KB block each)
    v16h bfr[4];
#pragma unroll
    for (int j = 0; j < 4; ++j) {
      int nt = wave * 4 + j;
      const _Float16* bp = w2p + ((size_t)(nt * 32 + ks) * 32 + lane) * 16;
      bfr[j] = *(const v16h*)bp;
      // prefetch next K slab (global_prefetch_b8); clamped to stay in bounds
      int ksn = (ks < 31) ? (ks + 1) : ks;
      __builtin_prefetch(w2p + ((size_t)(nt * 32 + ksn) * 32 + lane) * 16, 0, 1);
    }
#pragma unroll
    for (int mt = 0; mt < 4; ++mt) {
      // A fragment from LDS per 16-bit A layout:
      // lane<16: K = klo+0..7 and klo+16..23 ; lane>=16: +8 on both chunks
      const _Float16* hp =
          hsm + (mt * 16 + lane15) * ROWSTRIDE_H + klo + hiHalf * 8;
      v8h lo = *(const v8h*)hp;
      v8h hi = *(const v8h*)(hp + 16);
      v16h a = __builtin_shufflevector(lo, hi, 0, 1, 2, 3, 4, 5, 6, 7,
                                       8, 9, 10, 11, 12, 13, 14, 15);
#pragma unroll
      for (int j = 0; j < 4; ++j)
        acc[mt][j] = __builtin_amdgcn_wmma_f32_16x16x32_f16(
            false, a, false, bfr[j], (short)0, acc[mt][j], false, false);
    }
  }

  // ---------------- Store: D-tile layout -> out[M, 256] --------------------
#pragma unroll
  for (int mt = 0; mt < 4; ++mt) {
    int rowg = wgBase + mt * 16 + hiHalf * 8;
#pragma unroll
    for (int j = 0; j < 4; ++j) {
      int col = (wave * 4 + j) * 16 + lane15;
#pragma unroll
      for (int r = 0; r < 8; ++r)
        out[(size_t)(rowg + r) * E_DIM + col] = acc[mt][j][r];
    }
  }
}

// -------------------------------------------------------------------------
extern "C" void kernel_launch(void* const* d_in, const int* in_sizes, int n_in,
                              void* d_out, int out_size, void* d_ws, size_t ws_size,
                              hipStream_t stream) {
  (void)in_sizes; (void)n_in; (void)out_size; (void)ws_size;

  const float* x     = (const float*)d_in[0];   // [8,4096,256] f32
  const float* theta = (const float*)d_in[1];   // [8] f32
  const float* W1    = (const float*)d_in[2];   // [1024,8] f32
  const float* W2    = (const float*)d_in[3];   // [256,1024] f32
  float* out = (float*)d_out;                   // [8,4096,256] f32

  _Float16* w1p = (_Float16*)d_ws;                          // 64 KB
  _Float16* w2p = (_Float16*)((char*)d_ws + 64 * 1024);     // 512 KB

  pack_w1_kernel<<<8, 256, 0, stream>>>(W1, theta, w1p);
  pack_w2_kernel<<<64, 256, 0, stream>>>(W2, w2p);

  (void)hipFuncSetAttribute((const void*)ffq_main_kernel,
                            hipFuncAttributeMaxDynamicSharedMemorySize,
                            LDS_BYTES);
  ffq_main_kernel<<<M_TOTAL / 64, 128, LDS_BYTES, stream>>>(x, w1p, w2p, out);
}
